// DecoderLayer_37177236914552
// MI455X (gfx1250) — matrix operand order repaired and checked by reading, not compiled, not run on hardware
//
#include <hip/hip_runtime.h>
#include <hip/hip_bf16.h>
#include <stdint.h>

typedef __attribute__((ext_vector_type(16))) _Float16 v16h;
typedef __attribute__((ext_vector_type(8)))  _Float16 v8h;
typedef __attribute__((ext_vector_type(4)))  _Float16 v4h;
typedef __attribute__((ext_vector_type(8)))  float    v8f;

#define DMODEL 1024
#define HEADS  16
#define HDIM   64
#define FFND   4096
#define BATCH  2
#define SEQ    2048
#define MTOK   (BATCH * SEQ)

// ---------------------------------------------------------------------------
// CDNA5 async global->LDS copy (ASYNCcnt-tracked, no VGPR staging).
// LDS byte offset = low 32 bits of the generic __shared__ pointer (flat-LDS
// aperture keeps the workgroup-relative offset in addr[31:0]).
// ---------------------------------------------------------------------------
__device__ __forceinline__ void async_ld_b128(uint32_t ldsoff, const void* g) {
  asm volatile("global_load_async_to_lds_b128 %0, %1, off"
               :: "v"(ldsoff), "v"((unsigned long long)(uintptr_t)g)
               : "memory");
}
__device__ __forceinline__ void wait_async() {
  asm volatile("s_wait_asynccnt 0x0" ::: "memory");
}

// ---------------------------------------------------------------------------
// fragment helpers (wave32 WMMA; this code's contiguous k order, k = 16*(lane/16) + s)
// ---------------------------------------------------------------------------
__device__ __forceinline__ v16h mk16(v8h lo, v8h hi) {
  return __builtin_shufflevector(lo, hi, 0,1,2,3,4,5,6,7,8,9,10,11,12,13,14,15);
}
// A-frag 16x32 f16: lanes 0-15 -> K 0-15 ; lanes 16-31 -> K 16-31
__device__ __forceinline__ v16h load_afrag(const _Float16* rowPtr, int kb) {
  v8h lo = *(const v8h*)(rowPtr + 2*kb);
  v8h hi = *(const v8h*)(rowPtr + 2*kb + 8);
  return mk16(lo, hi);
}
// B-frag 32x16 f16: lanes 0-15 hold K=0-15, lanes 16-31 hold K=16-31
__device__ __forceinline__ v16h load_bfrag(const _Float16* colPtr) {
  v8h lo = *(const v8h*)(colPtr);
  v8h hi = *(const v8h*)(colPtr + 8);
  return mk16(lo, hi);
}

// ---------------------------------------------------------------------------
// f32 -> f16 conversion pass (one-shot; halves all downstream read traffic)
// ---------------------------------------------------------------------------
__global__ __launch_bounds__(256)
void cvt_f16_kernel(const float* __restrict__ in, _Float16* __restrict__ out) {
  const size_t i = ((size_t)blockIdx.x * 256 + threadIdx.x) * 8;
  const float4 a = *(const float4*)(in + i);
  const float4 b = *(const float4*)(in + i + 4);
  v8h o;
  o[0] = (_Float16)a.x; o[1] = (_Float16)a.y; o[2] = (_Float16)a.z; o[3] = (_Float16)a.w;
  o[4] = (_Float16)b.x; o[5] = (_Float16)b.y; o[6] = (_Float16)b.z; o[7] = (_Float16)b.w;
  *(v8h*)(out + i) = o;
}

// ---------------------------------------------------------------------------
// GEMM: C[M,N] = A[M,K] * W[K,N] + bias[N]  (optional ReLU)
// f16 in/out, f32 accum. A tile copied via async-to-LDS; W tile transposed.
// block = 256 threads (8 waves), tile 128x128, K-step 32
// ---------------------------------------------------------------------------
#define BM 128
#define BN 128
#define BKG 32
#define LDK 40   // 32 + 8 pad halves; row stride 80B (16B-aligned)

__global__ __launch_bounds__(256)
void gemm_bias_kernel(const _Float16* __restrict__ A, const _Float16* __restrict__ W,
                      const float* __restrict__ bias, _Float16* __restrict__ C,
                      int N, int K, int doRelu)
{
  __shared__ __align__(16) _Float16 lA[BM * LDK];
  __shared__ __align__(16) _Float16 lB[BN * LDK];

  const int tid  = threadIdx.x;
  const int lane = tid & 31;
  const int wid  = tid >> 5;
  const int wm   = wid >> 1;
  const int wn   = wid & 1;
  const int ln   = lane & 15;
  const int hi16 = lane >> 4;
  const int rowBase = blockIdx.y * BM;
  const int colBase = blockIdx.x * BN;
  const uint32_t lA0 = (uint32_t)(uintptr_t)(&lA[0]);

  v8f zero = {};
  v8f acc[2][4];
#pragma unroll
  for (int i = 0; i < 2; ++i)
#pragma unroll
    for (int j = 0; j < 4; ++j) acc[i][j] = zero;

  // A tile: 128 rows x 32 halves = 512 x 16B chunks; thread owns {tid, tid+256}
  const int aR0 = tid >> 2,         aC = (tid & 3) * 8;
  const int aR1 = (tid + 256) >> 2;
  // W tile: 32 k-rows x 128 halves = 512 x 8-half chunks; thread owns {tid, tid+256}
  const int wK0 = tid >> 4,         wN = (tid & 15) * 8;
  const int wK1 = (tid + 256) >> 4;

  for (int k0 = 0; k0 < K; k0 += BKG) {
    // ---- A tile: async byte-copy into LDS (row-major f16 [128][32]) ----
    async_ld_b128(lA0 + (uint32_t)(aR0 * LDK + aC) * 2,
                  A + (size_t)(rowBase + aR0) * K + k0 + aC);
    async_ld_b128(lA0 + (uint32_t)(aR1 * LDK + aC) * 2,
                  A + (size_t)(rowBase + aR1) * K + k0 + aC);
    // ---- W tile: f16 load + transpose into LDS [n][k] ----
    {
      const v8h w0 = *(const v8h*)(W + (size_t)(k0 + wK0) * N + colBase + wN);
      const v8h w1 = *(const v8h*)(W + (size_t)(k0 + wK1) * N + colBase + wN);
#pragma unroll
      for (int j = 0; j < 8; ++j) lB[(wN + j) * LDK + wK0] = w0[j];
#pragma unroll
      for (int j = 0; j < 8; ++j) lB[(wN + j) * LDK + wK1] = w1[j];
    }
    if (k0 + BKG < K)  // prefetch next W K-tile (global_prefetch_b8)
      __builtin_prefetch(W + (size_t)(k0 + BKG + wK0) * N + colBase + wN, 0, 1);

    wait_async();
    __syncthreads();

    const int kb = hi16 * 8;
    v16h afr[2];
#pragma unroll
    for (int mt = 0; mt < 2; ++mt)
      afr[mt] = load_afrag(lA + (wm*32 + mt*16 + ln) * LDK, kb);
#pragma unroll
    for (int nt = 0; nt < 4; ++nt) {
      v16h bfr = load_bfrag(lB + (wn*64 + nt*16 + ln) * LDK + hi16*16);
#pragma unroll
      for (int mt = 0; mt < 2; ++mt)
        acc[mt][nt] = __builtin_amdgcn_wmma_f32_16x16x32_f16(
            false, afr[mt], false, bfr, (short)0, acc[mt][nt], false, false);
    }
    __syncthreads();
  }

  // epilogue (C/D layout: VGPR r -> row r + 8*(lane>>4), col = lane&15)
#pragma unroll
  for (int nt = 0; nt < 4; ++nt) {
    const int col = colBase + wn*64 + nt*16 + ln;
    const float bv = bias[col];
#pragma unroll
    for (int mt = 0; mt < 2; ++mt) {
      const int row0 = rowBase + wm*32 + mt*16 + hi16*8;
#pragma unroll
      for (int r = 0; r < 8; ++r) {
        float v = acc[mt][nt][r] + bv;
        if (doRelu) v = fmaxf(v, 0.f);
        C[(size_t)(row0 + r) * N + col] = (_Float16)v;
      }
    }
  }
}

// ---------------------------------------------------------------------------
// Flash attention: one block = 128 q-rows of one (b,h); each wave owns 16
// full score rows. Q/K tiles via async-to-LDS; V transposed manually.
// Output written as [b, s, h, dh] in f16.
// ---------------------------------------------------------------------------
#define BQ  128
#define TKV 64
#define QLD 72   // 64 + 8 pad halves; row stride 144B (16B-aligned)

__global__ __launch_bounds__(256)
void flash_attn_kernel(const _Float16* __restrict__ Qb, int qStride, int qHead,
                       const _Float16* __restrict__ Kb, const _Float16* __restrict__ Vb,
                       int kvStride, int kvHead,
                       const float* __restrict__ mask, _Float16* __restrict__ Out,
                       float scale)
{
  __shared__ __align__(16) _Float16 sQ[BQ  * QLD];
  __shared__ __align__(16) _Float16 sK[TKV * QLD];
  __shared__ __align__(16) _Float16 sV[HDIM * QLD];  // transposed [d][kv]
  __shared__ __align__(16) _Float16 sP[BQ  * QLD];

  const int tid  = threadIdx.x;
  const int lane = tid & 31;
  const int wid  = tid >> 5;
  const int ln   = lane & 15;
  const int hi16 = lane >> 4;
  const int h  = blockIdx.y;
  const int b  = blockIdx.z;
  const int q0 = blockIdx.x * BQ;
  const size_t tokBase = (size_t)b * SEQ;
  const uint32_t sQ0 = (uint32_t)(uintptr_t)(&sQ[0]);
  const uint32_t sK0 = (uint32_t)(uintptr_t)(&sK[0]);

  // ---- Q tile 128x64 halves: 1024 x 16B chunks -> 4 async copies/thread ----
#pragma unroll
  for (int i = 0; i < 4; ++i) {
    const int ch = tid + 256 * i;
    const int r = ch >> 3, c8 = (ch & 7) * 8;
    async_ld_b128(sQ0 + (uint32_t)(r * QLD + c8) * 2,
                  Qb + (tokBase + q0 + r) * (size_t)qStride + h * qHead + c8);
  }

  float rm[8], rl[8];
  v8f o[4];
  v8f zero = {};
#pragma unroll
  for (int r = 0; r < 8; ++r) { rm[r] = -1e30f; rl[r] = 0.f; }
#pragma unroll
  for (int d = 0; d < 4; ++d) o[d] = zero;

  // K/V tiles: 64 rows x 64 halves = 512 x 8-half chunks; thread owns {tid, tid+256}
  const int kR0 = tid >> 3, kC = (tid & 7) * 8;
  const int kR1 = (tid + 256) >> 3;

  for (int t = 0; t < SEQ; t += TKV) {
    // K tile: async byte-copy [kv][d]
    async_ld_b128(sK0 + (uint32_t)(kR0 * QLD + kC) * 2,
                  Kb + (tokBase + t + kR0) * (size_t)kvStride + h * kvHead + kC);
    async_ld_b128(sK0 + (uint32_t)(kR1 * QLD + kC) * 2,
                  Kb + (tokBase + t + kR1) * (size_t)kvStride + h * kvHead + kC);
    // V tile: f16 load + transpose -> [d][kv]
    {
      const v8h v0 = *(const v8h*)(Vb + (tokBase + t + kR0) * (size_t)kvStride + h * kvHead + kC);
      const v8h v1 = *(const v8h*)(Vb + (tokBase + t + kR1) * (size_t)kvStride + h * kvHead + kC);
#pragma unroll
      for (int j = 0; j < 8; ++j) sV[(kC + j) * QLD + kR0] = v0[j];
#pragma unroll
      for (int j = 0; j < 8; ++j) sV[(kC + j) * QLD + kR1] = v1[j];
    }
    wait_async();
    __syncthreads();

    // ---- S = Q @ K^T : 16x64 per wave ----
    const int kb = hi16 * 8;
    const _Float16* qrow = sQ + (wid*16 + ln) * QLD;
    v16h qa0 = load_afrag(qrow,      kb);
    v16h qa1 = load_afrag(qrow + 32, kb);
    v8f s[4];
#pragma unroll
    for (int nt = 0; nt < 4; ++nt) {
      const _Float16* kp = sK + (nt*16 + ln) * QLD + hi16*16;
      v16h b0 = load_bfrag(kp);
      v16h b1 = load_bfrag(kp + 32);
      v8f a = zero;
      a = __builtin_amdgcn_wmma_f32_16x16x32_f16(false, qa0, false, b0, (short)0, a, false, false);
      a = __builtin_amdgcn_wmma_f32_16x16x32_f16(false, qa1, false, b1, (short)0, a, false, false);
      s[nt] = a;
    }

    // scale + additive mask
    const int qrow0 = q0 + wid*16 + hi16*8;
#pragma unroll
    for (int nt = 0; nt < 4; ++nt) {
      const float* mp = mask + (size_t)qrow0 * SEQ + t + nt*16 + ln;
#pragma unroll
      for (int r = 0; r < 8; ++r) s[nt][r] = s[nt][r] * scale + mp[(size_t)r * SEQ];
    }

    // ---- online softmax over this wave's 16 rows ----
    float al[8];
#pragma unroll
    for (int r = 0; r < 8; ++r) {
      float mx = fmaxf(fmaxf(s[0][r], s[1][r]), fmaxf(s[2][r], s[3][r]));
      mx = fmaxf(mx, __shfl_xor(mx, 1));
      mx = fmaxf(mx, __shfl_xor(mx, 2));
      mx = fmaxf(mx, __shfl_xor(mx, 4));
      mx = fmaxf(mx, __shfl_xor(mx, 8));
      const float mnew = fmaxf(rm[r], mx);
      const float alpha = __expf(rm[r] - mnew);
      rm[r] = mnew;
      float ps = 0.f;
#pragma unroll
      for (int nt = 0; nt < 4; ++nt) {
        float p = __expf(s[nt][r] - mnew);
        s[nt][r] = p;
        ps += p;
      }
      ps += __shfl_xor(ps, 1);
      ps += __shfl_xor(ps, 2);
      ps += __shfl_xor(ps, 4);
      ps += __shfl_xor(ps, 8);
      rl[r] = rl[r] * alpha + ps;
      al[r] = alpha;
    }
#pragma unroll
    for (int d = 0; d < 4; ++d)
#pragma unroll
      for (int r = 0; r < 8; ++r) o[d][r] *= al[r];

    // ---- restage P (C-layout -> LDS A-layout staging, same-wave region) ----
    _Float16* prow = sP + (wid*16 + hi16*8) * QLD + ln;
#pragma unroll
    for (int nt = 0; nt < 4; ++nt)
#pragma unroll
      for (int r = 0; r < 8; ++r) prow[r * QLD + nt*16] = (_Float16)s[nt][r];

    // ---- O += P @ V ----
    const _Float16* pr = sP + (wid*16 + ln) * QLD;
    v16h pa0 = load_afrag(pr,      kb);
    v16h pa1 = load_afrag(pr + 32, kb);
#pragma unroll
    for (int dt = 0; dt < 4; ++dt) {
      const _Float16* vp = sV + (dt*16 + ln) * QLD + hi16*16;
      v16h b0 = load_bfrag(vp);
      v16h b1 = load_bfrag(vp + 32);
      o[dt] = __builtin_amdgcn_wmma_f32_16x16x32_f16(false, pa0, false, b0, (short)0, o[dt], false, false);
      o[dt] = __builtin_amdgcn_wmma_f32_16x16x32_f16(false, pa1, false, b1, (short)0, o[dt], false, false);
    }
    __syncthreads();
  }

  // ---- write O / l as f16 [b, s, h, dh] ----
  const int orow0 = q0 + wid*16 + hi16*8;
#pragma unroll
  for (int dt = 0; dt < 4; ++dt) {
    const int col = h * HDIM + dt*16 + ln;
#pragma unroll
    for (int r = 0; r < 8; ++r)
      Out[(tokBase + orow0 + r) * (size_t)DMODEL + col] = (_Float16)(o[dt][r] / rl[r]);
  }
}

// ---------------------------------------------------------------------------
// LayerNorm(a + res): f16 inputs, f16 intermediate output or f32 final output
// ---------------------------------------------------------------------------
__global__ __launch_bounds__(256)
void add_layernorm_kernel(const _Float16* __restrict__ a, const _Float16* __restrict__ res,
                          const float* __restrict__ gamma, const float* __restrict__ beta,
                          _Float16* __restrict__ outH, float* __restrict__ outF)
{
  __shared__ float sred[16];
  const int tid = threadIdx.x;
  const size_t row = blockIdx.x;
  const v4h va = ((const v4h*)(a   + row * DMODEL))[tid];
  const v4h vr = ((const v4h*)(res + row * DMODEL))[tid];
  const float x0 = (float)va[0] + (float)vr[0];
  const float x1 = (float)va[1] + (float)vr[1];
  const float x2 = (float)va[2] + (float)vr[2];
  const float x3 = (float)va[3] + (float)vr[3];
  float s = x0 + x1 + x2 + x3;
  float q = x0*x0 + x1*x1 + x2*x2 + x3*x3;
#pragma unroll
  for (int m = 1; m < 32; m <<= 1) { s += __shfl_xor(s, m); q += __shfl_xor(q, m); }
  const int lane = tid & 31, wid = tid >> 5;
  if (lane == 0) { sred[wid] = s; sred[8 + wid] = q; }
  __syncthreads();
  if (tid < 2) {
    float acc = 0.f;
#pragma unroll
    for (int i = 0; i < 8; ++i) acc += sred[tid*8 + i];
    sred[tid*8] = acc;
  }
  __syncthreads();
  const float mean = sred[0] * (1.0f / DMODEL);
  const float var  = sred[8] * (1.0f / DMODEL) - mean * mean;
  const float rstd = rsqrtf(var + 1e-5f);
  const float4 g4 = ((const float4*)gamma)[tid];
  const float4 b4 = ((const float4*)beta)[tid];
  const float r0 = g4.x * (x0 - mean) * rstd + b4.x;
  const float r1 = g4.y * (x1 - mean) * rstd + b4.y;
  const float r2 = g4.z * (x2 - mean) * rstd + b4.z;
  const float r3 = g4.w * (x3 - mean) * rstd + b4.w;
  if (outF) {
    float4 o4; o4.x = r0; o4.y = r1; o4.z = r2; o4.w = r3;
    ((float4*)(outF + row * DMODEL))[tid] = o4;
  } else {
    v4h oh;
    oh[0] = (_Float16)r0; oh[1] = (_Float16)r1; oh[2] = (_Float16)r2; oh[3] = (_Float16)r3;
    ((v4h*)(outH + row * DMODEL))[tid] = oh;
  }
}

// ---------------------------------------------------------------------------
// host-side orchestration
// ---------------------------------------------------------------------------
extern "C" void kernel_launch(void* const* d_in, const int* in_sizes, int n_in,
                              void* d_out, int out_size, void* d_ws, size_t ws_size,
                              hipStream_t stream) {
  (void)in_sizes; (void)n_in; (void)out_size; (void)ws_size;
  const float* x     = (const float*)d_in[0];
  const float* y     = (const float*)d_in[1];
  const float* smask = (const float*)d_in[2];
  const float* cmask = (const float*)d_in[3];
  const float* Wqkv  = (const float*)d_in[4];
  const float* bqkv  = (const float*)d_in[5];
  const float* Wo1   = (const float*)d_in[6];
  const float* bo1   = (const float*)d_in[7];
  const float* Wkv   = (const float*)d_in[8];
  const float* bkv   = (const float*)d_in[9];
  const float* Wq    = (const float*)d_in[10];
  const float* bq    = (const float*)d_in[11];
  const float* Wo2   = (const float*)d_in[12];
  const float* bo2   = (const float*)d_in[13];
  const float* g1    = (const float*)d_in[14];
  const float* b1    = (const float*)d_in[15];
  const float* g2    = (const float*)d_in[16];
  const float* b2    = (const float*)d_in[17];
  const float* g3    = (const float*)d_in[18];
  const float* b3    = (const float*)d_in[19];
  const float* W1    = (const float*)d_in[20];
  const float* bw1   = (const float*)d_in[21];
  const float* W2    = (const float*)d_in[22];
  const float* bw2   = (const float*)d_in[23];
  float* out = (float*)d_out;
  _Float16* hw = (_Float16*)d_ws;

  // workspace layout (halves); hbuf overlays qkv (dead by FFN stage)
  _Float16* x16    = hw;                  // 4,194,304
  _Float16* y16    = hw + 4194304;        // 4,194,304
  _Float16* wqkv16 = hw + 8388608;        // 3,145,728
  _Float16* wo116  = hw + 11534336;       // 1,048,576
  _Float16* wkv16  = hw + 12582912;       // 2,097,152
  _Float16* wq16   = hw + 14680064;       // 1,048,576
  _Float16* wo216  = hw + 15728640;       // 1,048,576
  _Float16* w116   = hw + 16777216;       // 4,194,304
  _Float16* w216   = hw + 20971520;       // 4,194,304
  _Float16* qkv16  = hw + 25165824;       // 12,582,912
  _Float16* hbuf   = hw + 25165824;       // 16,777,216 (reuse, ends at kv16)
  _Float16* kv16   = hw + 41943040;       // 8,388,608
  _Float16* q216   = hw + 50331648;       // 4,194,304
  _Float16* vals16 = hw + 54525952;       // 4,194,304
  _Float16* tmp16  = hw + 58720256;       // 4,194,304
  _Float16* y116   = hw + 62914560;       // 4,194,304
  _Float16* y216   = hw + 67108864;       // 4,194,304

  const dim3 blk(256);
  const float scale = 0.125f; // 1/sqrt(64)
  const dim3 attnGrid(SEQ / BQ, HEADS, BATCH);

  // 0. one-shot f32 -> f16 conversions (2048 elems per block)
  cvt_f16_kernel<<<dim3(2048), blk, 0, stream>>>(x,    x16);
  cvt_f16_kernel<<<dim3(2048), blk, 0, stream>>>(y,    y16);
  cvt_f16_kernel<<<dim3(1536), blk, 0, stream>>>(Wqkv, wqkv16);
  cvt_f16_kernel<<<dim3(512),  blk, 0, stream>>>(Wo1,  wo116);
  cvt_f16_kernel<<<dim3(1024), blk, 0, stream>>>(Wkv,  wkv16);
  cvt_f16_kernel<<<dim3(512),  blk, 0, stream>>>(Wq,   wq16);
  cvt_f16_kernel<<<dim3(512),  blk, 0, stream>>>(Wo2,  wo216);
  cvt_f16_kernel<<<dim3(2048), blk, 0, stream>>>(W1,   w116);
  cvt_f16_kernel<<<dim3(2048), blk, 0, stream>>>(W2,   w216);

  // 1. qkv = y @ Wqkv + bqkv            [4096 x 3072]
  gemm_bias_kernel<<<dim3(3072/BN, MTOK/BM), blk, 0, stream>>>(y16, wqkv16, bqkv, qkv16, 3072, DMODEL, 0);
  // 2. self flash-attention -> vals [b,s,h,dh]
  flash_attn_kernel<<<attnGrid, blk, 0, stream>>>(qkv16, 3072, 192, qkv16 + 64, qkv16 + 128,
                                                  3072, 192, smask, vals16, scale);
  // 3. attn_out = vals @ Wo1 + bo1
  gemm_bias_kernel<<<dim3(1024/BN, MTOK/BM), blk, 0, stream>>>(vals16, wo116, bo1, tmp16, 1024, DMODEL, 0);
  // 4. y1 = LN(attn_out + y)
  add_layernorm_kernel<<<dim3(MTOK), blk, 0, stream>>>(tmp16, y16, g1, b1, y116, nullptr);
  // 5. kv = x @ Wkv + bkv               [4096 x 2048]
  gemm_bias_kernel<<<dim3(2048/BN, MTOK/BM), blk, 0, stream>>>(x16, wkv16, bkv, kv16, 2048, DMODEL, 0);
  // 6. q2 = y1 @ Wq + bq
  gemm_bias_kernel<<<dim3(1024/BN, MTOK/BM), blk, 0, stream>>>(y116, wq16, bq, q216, 1024, DMODEL, 0);
  // 7. cross flash-attention -> vals
  flash_attn_kernel<<<attnGrid, blk, 0, stream>>>(q216, 1024, 64, kv16, kv16 + 64,
                                                  2048, 128, cmask, vals16, scale);
  // 8. cross_out = vals @ Wo2 + bo2
  gemm_bias_kernel<<<dim3(1024/BN, MTOK/BM), blk, 0, stream>>>(vals16, wo216, bo2, tmp16, 1024, DMODEL, 0);
  // 9. y2 = LN(cross_out + y1)
  add_layernorm_kernel<<<dim3(MTOK), blk, 0, stream>>>(tmp16, y116, g2, b2, y216, nullptr);
  // 10. h = relu(y2 @ W1 + bw1)         [4096 x 4096]
  gemm_bias_kernel<<<dim3(4096/BN, MTOK/BM), blk, 0, stream>>>(y216, w116, bw1, hbuf, FFND, DMODEL, 1);
  // 11. ffn = h @ W2 + bw2
  gemm_bias_kernel<<<dim3(1024/BN, MTOK/BM), blk, 0, stream>>>(hbuf, w216, bw2, tmp16, 1024, FFND, 0);
  // 12. out = LN(ffn + y2) -> f32 d_out
  add_layernorm_kernel<<<dim3(MTOK), blk, 0, stream>>>(tmp16, y216, g3, b3, nullptr, out);
}
